// pointer_28578712388353
// MI455X (gfx1250) — matrix operand order repaired and checked by reading, not compiled, not run on hardware
//
#include <hip/hip_runtime.h>
#include <hip/hip_bf16.h>

// v16h = 16 halves (A/B operand of v_wmma_f32_16x16x32_f16), v8f = f32 accum tile
typedef __attribute__((ext_vector_type(16))) _Float16 v16h;
typedef __attribute__((ext_vector_type(8)))  float    v8f;

#define S_LEN 65536

// ---------------------------------------------------------------------------
// Branch-free transcendental helpers.  gfx1250 has a hardware v_tanh_f32
// (TRANS32 pipe, co-executes with WMMA); fall back to a clamped exp form.
// ---------------------------------------------------------------------------
__device__ __forceinline__ float fast_tanh(float x) {
#if __has_builtin(__builtin_amdgcn_tanhf)
  return __builtin_amdgcn_tanhf(x);
#else
  x = fminf(fmaxf(x, -15.0f), 15.0f);
  float e = __expf(2.0f * x);
  return (e - 1.0f) / (e + 1.0f);
#endif
}

__device__ __forceinline__ float sigf(float x) {
  return 1.0f / (1.0f + __expf(-x));
}

// ---------------------------------------------------------------------------
// Kernel 1: LSTM step + d_proj + fused bias, and pack W_enc -> f16 B-operand
// layout.  B layout per WMMA tile (32x16, f16): lane n = N column (lane&15),
// halves j=0..15 cover K = ck*32 + 8*(lane>>4) + (j&7) + 16*(j>>3).
// ---------------------------------------------------------------------------
__global__ __launch_bounds__(256) void prep_kernel(
    const float* __restrict__ x,      // start_token [256]
    const float* __restrict__ h0,     // [256]
    const float* __restrict__ c0,     // [256]
    const float* __restrict__ Kmat,   // lstm_kernel  [256,1024]
    const float* __restrict__ Rmat,   // lstm_rkernel [256,1024]
    const float* __restrict__ lbias,  // lstm_bias [1024]
    const float* __restrict__ Wenc,   // [256,256]
    const float* __restrict__ benc,   // [256]
    const float* __restrict__ Wdec,   // [256,256]
    const float* __restrict__ bdec,   // [256]
    float* __restrict__ biascomb,     // out: [256]  b_enc + d_proj
    _Float16* __restrict__ Bp)        // out: [65536] packed W_enc f16
{
  __shared__ float z[1024];
  __shared__ float hv[256];
  const int tid = threadIdx.x;

  // z = x @ K + h0 @ R + bias   (each thread: 4 of the 1024 gate outputs)
  #pragma unroll
  for (int q = 0; q < 4; ++q) {
    int g = tid + q * 256;
    float acc = lbias[g];
    for (int d = 0; d < 256; ++d) acc += x[d]  * Kmat[d * 1024 + g];
    for (int d = 0; d < 256; ++d) acc += h0[d] * Rmat[d * 1024 + g];
    z[g] = acc;
  }
  __syncthreads();

  // gates (order i, f, g, o)
  {
    int j = tid;
    float i_ = z[j], f_ = z[j + 256], g_ = z[j + 512], o_ = z[j + 768];
    float c  = sigf(f_) * c0[j] + sigf(i_) * fast_tanh(g_);
    hv[j]    = sigf(o_) * fast_tanh(c);
  }
  __syncthreads();

  // d_proj = h @ W_dec + b_dec ; fuse with b_enc
  {
    int j = tid;
    float acc = bdec[j];
    for (int k = 0; k < 256; ++k) acc += hv[k] * Wdec[k * 256 + j];
    biascomb[j] = acc + benc[j];
  }

  // Pack W_enc into WMMA-B tile order: chunk (cn,ck) is a 32x16 f16 tile,
  // stored as 512 halves: [lane][j] with K = ck*32 + 8*(lane>>4) + (j&7) + 16*(j>>3),
  //                                   N = cn*16 + (lane&15).
  for (int t = tid; t < 65536; t += 256) {
    int chunk  = t >> 9;
    int within = t & 511;
    int lane   = within >> 4;
    int j      = within & 15;
    int cn     = chunk >> 3;
    int ck     = chunk & 7;
    int Kidx   = ck * 32 + ((lane >> 4) << 3) + (j & 7) + ((j >> 3) << 4);
    int N      = (cn << 4) + (lane & 15);
    Bp[t] = (_Float16)Wenc[Kidx * 256 + N];
  }
}

// ---------------------------------------------------------------------------
// Kernel 2: u[s] = tanh(E[s,:] @ W_enc + biascomb) . v_w + v_b   via WMMA.
// Grid: 512 blocks x 256 threads (8 waves).  Wave w owns rows
// [blockIdx*128 + w*16, +16).  A kept fully in registers (K=256 -> 8 x v16h).
// Column-tile loop fully unrolled: 128 static v_wmma per wave.
// ---------------------------------------------------------------------------
__global__ __launch_bounds__(256) void attn_gemm_kernel(
    const float*    __restrict__ enc,       // [65536,256] f32
    const _Float16* __restrict__ Bp,        // packed W_enc f16
    const float*    __restrict__ biascomb,  // [256]
    const float*    __restrict__ vw,        // [256]
    const float*    __restrict__ vb,        // [1]
    float*          __restrict__ U)         // [65536] logits
{
  const int lane = threadIdx.x & 31;
  const int wave = threadIdx.x >> 5;
  const int hi   = lane >> 4;     // half-wave selector
  const int m    = lane & 15;     // row within tile (A) / col within tile (B,C)
  const int row0 = blockIdx.x * 128 + wave * 16;

  // --- load A operands: 16 rows x K=256, f32 -> f16, ISA A-layout ----------
  // lanes 0-15:  halves 0..7 = K ck*32+0..7,   halves 8..15 = K ck*32+16..23
  // lanes 16-31: halves 0..7 = K ck*32+8..15,  halves 8..15 = K ck*32+24..31
  const float* Erow = enc + (size_t)(row0 + m) * 256;
  v16h A[8];
  #pragma unroll
  for (int ck = 0; ck < 8; ++ck) {
    const int kb = ck * 32 + 8 * hi;
    #pragma unroll
    for (int j = 0; j < 8; ++j) A[ck][j]     = (_Float16)Erow[kb + j];
    #pragma unroll
    for (int j = 0; j < 8; ++j) A[ck][8 + j] = (_Float16)Erow[kb + 16 + j];
  }

  float rowacc[8];
  #pragma unroll
  for (int r = 0; r < 8; ++r) rowacc[r] = 0.0f;

  // --- 16 column tiles, each K=256 in 8 WMMA steps; fuse tanh + v_w --------
  #pragma unroll
  for (int cn = 0; cn < 16; ++cn) {
    v8f acc = {};
    const v16h* bp = (const v16h*)(Bp + ((cn * 8) << 9)) + lane;
    #pragma unroll
    for (int ck = 0; ck < 8; ++ck) {
      v16h b = bp[ck * 32];   // 32B aligned per-lane load of packed B tile
      acc = __builtin_amdgcn_wmma_f32_16x16x32_f16(
          /*neg_a=*/false, A[ck], /*neg_b=*/false, b,
          /*c_mod=*/(short)0, acc, /*reuse_a=*/false, /*reuse_b=*/false);
    }
    const int n    = cn * 16 + m;
    const float bn = biascomb[n];
    const float wn = vw[n];
    // C layout: VGPR r holds row (r + 8*hi), column = m
    #pragma unroll
    for (int r = 0; r < 8; ++r) rowacc[r] += fast_tanh(acc[r] + bn) * wn;
  }

  // --- reduce over 16 columns held across the 16 lanes of each half --------
  #pragma unroll
  for (int r = 0; r < 8; ++r) {
    float v = rowacc[r];
    v += __shfl_xor(v, 1);
    v += __shfl_xor(v, 2);
    v += __shfl_xor(v, 4);
    v += __shfl_xor(v, 8);
    rowacc[r] = v;
  }
  if (m == 0) {
    const float b0 = vb[0];
    #pragma unroll
    for (int r = 0; r < 8; ++r) U[row0 + 8 * hi + r] = rowacc[r] + b0;
  }
}

// ---------------------------------------------------------------------------
// Kernel 3: single-block softmax pass 1: global max, exp, sum -> out, stats
// ---------------------------------------------------------------------------
__global__ __launch_bounds__(1024) void softmax_pass1(
    const float* __restrict__ U, float* __restrict__ out, float* __restrict__ stats)
{
  __shared__ float red[1024];
  const int tid = threadIdx.x;

  float mx = -1e30f;
  for (int i = tid; i < S_LEN; i += 1024) mx = fmaxf(mx, U[i]);
  red[tid] = mx;
  __syncthreads();
  for (int s = 512; s > 0; s >>= 1) {
    if (tid < s) red[tid] = fmaxf(red[tid], red[tid + s]);
    __syncthreads();
  }
  const float gmax = red[0];
  __syncthreads();

  float sum = 0.0f;
  for (int i = tid; i < S_LEN; i += 1024) {
    float e = __expf(U[i] - gmax);   // branch-free v_exp_f32 path
    out[i] = e;
    sum += e;
  }
  red[tid] = sum;
  __syncthreads();
  for (int s = 512; s > 0; s >>= 1) {
    if (tid < s) red[tid] += red[tid + s];
    __syncthreads();
  }
  if (tid == 0) stats[0] = 1.0f / red[0];
}

// Kernel 4: scale by 1/sum
__global__ __launch_bounds__(1024) void softmax_pass2(
    const float* __restrict__ stats, float* __restrict__ out)
{
  const int i = blockIdx.x * blockDim.x + threadIdx.x;
  if (i < S_LEN) out[i] *= stats[0];
}

// ---------------------------------------------------------------------------
extern "C" void kernel_launch(void* const* d_in, const int* in_sizes, int n_in,
                              void* d_out, int out_size, void* d_ws, size_t ws_size,
                              hipStream_t stream) {
  const float* enc  = (const float*)d_in[0];   // encoder_output [1,S,256]
  const float* h0   = (const float*)d_in[1];
  const float* c0   = (const float*)d_in[2];
  const float* x    = (const float*)d_in[3];   // start_token
  const float* K    = (const float*)d_in[4];   // lstm_kernel
  const float* R    = (const float*)d_in[5];   // lstm_rkernel
  const float* lb   = (const float*)d_in[6];   // lstm_bias
  const float* Wenc = (const float*)d_in[7];
  const float* benc = (const float*)d_in[8];
  const float* Wdec = (const float*)d_in[9];
  const float* bdec = (const float*)d_in[10];
  const float* vw   = (const float*)d_in[11];
  const float* vb   = (const float*)d_in[12];
  float* out = (float*)d_out;

  char* ws = (char*)d_ws;
  float*    biascomb = (float*)ws;                       // 1 KB
  float*    stats    = (float*)(ws + 1024);              // 4 B
  _Float16* Bp       = (_Float16*)(ws + 2048);           // 128 KB (32B aligned)
  float*    U        = (float*)(ws + 2048 + 131072);     // 256 KB

  hipLaunchKernelGGL(prep_kernel, dim3(1), dim3(256), 0, stream,
                     x, h0, c0, K, R, lb, Wenc, benc, Wdec, bdec, biascomb, Bp);
  hipLaunchKernelGGL(attn_gemm_kernel, dim3(512), dim3(256), 0, stream,
                     enc, Bp, biascomb, vw, vb, U);
  hipLaunchKernelGGL(softmax_pass1, dim3(1), dim3(1024), 0, stream, U, out, stats);
  hipLaunchKernelGGL(softmax_pass2, dim3(64), dim3(1024), 0, stream, stats, out);
}
